// NTU_29600914604805
// MI455X (gfx1250) — matrix operand order repaired and checked by reading, not compiled, not run on hardware
//
#include <hip/hip_runtime.h>
#include <hip/hip_bf16.h>
#include <cstdint>
#include <cstddef>

typedef __attribute__((ext_vector_type(16))) _Float16 v16h;
typedef __attribute__((ext_vector_type(8)))  _Float16 v8h;
typedef __attribute__((ext_vector_type(8)))  float    v8f;

#define NBR_K 9
#define MAXC  12   // max cs-chunks per wave: cs_p16<=1536 -> 96 chunks / 8 waves

// LDS byte offset of a generic pointer that points into shared memory.
// addrspace(3) pointers on AMDGPU are 32-bit offsets from the LDS base.
__device__ __forceinline__ uint32_t lds_offset_of(const void* p) {
  return (uint32_t)(uintptr_t)(__attribute__((address_space(3))) const char*)p;
}

// Async DMA memory -> LDS (16B per lane), tracked by ASYNCcnt (gfx1250).
__device__ __forceinline__ void async_load_b128_to_lds(uint32_t lds_off, uint32_t goff,
                                                       const void* sbase) {
  asm volatile("global_load_async_to_lds_b128 %0, %1, %2"
               :: "v"(lds_off), "v"(goff), "s"(sbase)
               : "memory");
}
__device__ __forceinline__ void wait_asynccnt0() {
  asm volatile("s_wait_asynccnt 0x0" ::: "memory");
}

// A-fragment loader, 16-bit A 16x32, contiguous k order (not the ISA 7.12.2 table order):
// lanes 0-15 hold K=k0+0..15 ; lanes 16-31 hold K=k0+16..31 (contiguous, matches B)
__device__ __forceinline__ v16h load_frag_a(const _Float16* base, int ld, int row,
                                            int k0, int klane) {
  const _Float16* p = base + (size_t)row * ld + k0 + 16 * klane;
  v8h lo = *(const v8h*)(p);
  v8h hi = *(const v8h*)(p + 8);
  return __builtin_shufflevector(lo, hi, 0,1,2,3,4,5,6,7,8,9,10,11,12,13,14,15);
}

__global__ __launch_bounds__(256) void pst_fused_kernel(
    const float* __restrict__ xys,      // (B,T_in,N,3)
    const float* __restrict__ feats,    // (B,T_in,N,cin) or null
    const _Float16* __restrict__ WsT,   // (cs_p16, din_p) transposed f16
    const float* __restrict__ bs, const float* __restrict__ gs, const float* __restrict__ Bs,
    const _Float16* __restrict__ WtT,   // (ct, cs_p2) transposed f16, this w
    float* __restrict__ acc,            // (B,T_out,A,ct)
    float* __restrict__ anchors_out,    // (B,T_out,A,3)
    int T_in, int N, int A, int T_out,
    int cin, int cs, int cs_p16, int cs_p2, int din_p, int ct,
    int tr, int ts, int samp, int dt, float r2, int accumulate)
{
  const int tid   = threadIdx.x;
  const int wave  = tid >> 5;
  const int lane  = tid & 31;
  const int klane = lane >> 4;
  const int lrow  = lane & 15;

  const int a0 = blockIdx.x * 16;
  const int tp = blockIdx.y;
  const int b  = blockIdx.z;
  const int t_anchor = tr + tp * ts;
  const int tfull    = t_anchor + dt;

  extern __shared__ char smem[];
  float*    ptsS = (float*)smem;                 // N*3 floats
  float*    ancS = ptsS + (size_t)N * 3;         // 48 floats
  int*      nbrS = (int*)(ancS + 48);            // 16*9 ints
  _Float16* sS   = (_Float16*)(nbrS + 144);      // 16*cs_p2 halves (16B aligned)
  _Float16* xS   = sS + (size_t)16 * cs_p2;      // 16*din_p halves

  // ---- stage points of frame tfull into LDS via async DMA (ASYNCcnt path) ----
  const float* ptsG = xys + ((size_t)b * T_in + tfull) * N * 3;
  {
    const uint32_t ldsBase = lds_offset_of(ptsS);
    const int nch = (N * 12) >> 4;               // 16B chunks; N*12 % 16 == 0
    for (int i = tid; i < nch; i += 256)
      async_load_b128_to_lds(ldsBase + (uint32_t)(i << 4), (uint32_t)(i << 4), ptsG);
    wait_asynccnt0();
  }
  // anchors (subsampled from anchor frame) + write-out for next layer (once, w==0)
  if (tid < 48) {
    int r = tid / 3, c = tid - r * 3;
    float v = xys[(((size_t)b * T_in + t_anchor) * N + (size_t)(a0 + r) * samp) * 3 + c];
    ancS[tid] = v;
    if (!accumulate)
      anchors_out[(((size_t)b * T_out + tp) * A + a0 + r) * 3 + c] = v;
  }
  // zero the s tile (covers zero-padded K rows for gemm2)
  for (int i = tid; i < 16 * cs_p2; i += 256) sS[i] = (_Float16)0.f;
  __syncthreads();

  // ---- ball query: first K hits in ascending index order, fill with first (or 0) ----
  if (tid < 16) {
    float ax = ancS[tid * 3 + 0], ay = ancS[tid * 3 + 1], az = ancS[tid * 3 + 2];
    int cnt = 0; int loc[NBR_K];
    for (int n = 0; n < N; ++n) {
      float dx = ptsS[n * 3 + 0] - ax;
      float dy = ptsS[n * 3 + 1] - ay;
      float dz = ptsS[n * 3 + 2] - az;
      if (dx * dx + dy * dy + dz * dz < r2) {
        loc[cnt++] = n;
        if (cnt == NBR_K) break;
      }
    }
    int fill = (cnt > 0) ? loc[0] : 0;
    for (int k2 = cnt; k2 < NBR_K; ++k2) loc[k2] = fill;
    for (int k2 = 0; k2 < NBR_K; ++k2) nbrS[tid * NBR_K + k2] = loc[k2];
  }

  // ---- per-neighbor MLP via WMMA, affine+ReLU epilogue, max over K in registers ----
  v8f zero = {};
  v8f smax[MAXC];
  #pragma unroll
  for (int i = 0; i < MAXC; ++i) smax[i] = zero;

  const int nKC = din_p >> 5;
  for (int k = 0; k < NBR_K; ++k) {
    __syncthreads();                       // previous gemm finished reading xS
    // build X tile: [disp(3) | gathered feats(cin) | zero pad] as f16
    for (int e = tid; e < 16 * din_p; e += 256) {
      int r = e / din_p, c = e - r * din_p;
      int idx = nbrS[r * NBR_K + k];
      float v = 0.f;
      if (c < 3)            v = ptsS[idx * 3 + c] - ancS[r * 3 + c];
      else if (c < 3 + cin) v = feats[(((size_t)b * T_in + tfull) * N + idx) * cin + (c - 3)];
      xS[e] = (_Float16)v;
    }
    __syncthreads();
    // (16 x din_p) @ (din_p x cs)
    for (int cc = wave, li = 0; cc * 16 < cs_p16; cc += 8, ++li) {
      int col = cc * 16 + lrow;
      v8f c8 = {};
      for (int kk = 0; kk < nKC; ++kk) {
        v16h af = load_frag_a(xS, din_p, lrow, kk * 32, klane);
        v16h bf = *(const v16h*)(WsT + (size_t)col * din_p + kk * 32 + klane * 16);
        c8 = __builtin_amdgcn_wmma_f32_16x16x32_f16(false, af, false, bf,
                                                    (short)0, c8, false, false);
      }
      float bsv = (col < cs) ? bs[col] : 0.f;
      float gsv = (col < cs) ? gs[col] : 0.f;
      float Bsv = (col < cs) ? Bs[col] : 0.f;
      #pragma unroll
      for (int j = 0; j < 8; ++j) {
        float h = (c8[j] + bsv) * gsv + Bsv;
        h = h > 0.f ? h : 0.f;
        smax[li][j] = smax[li][j] > h ? smax[li][j] : h;
      }
    }
  }
  __syncthreads();
  // drop s tile to LDS as f16 (columns >= cs stay zero)
  for (int cc = wave, li = 0; cc * 16 < cs_p16; cc += 8, ++li) {
    int col = cc * 16 + lrow;
    if (col < cs) {
      #pragma unroll
      for (int j = 0; j < 8; ++j)
        sS[(size_t)(j + 8 * klane) * cs_p2 + col] = (_Float16)smax[li][j];
    }
  }
  __syncthreads();

  // ---- temporal gemm: acc(16 x ct) (+)= s(16 x cs) @ Wt[w](cs x ct) ----
  const int nKC2 = cs_p2 >> 5;
  const size_t accBase = ((size_t)b * T_out + tp) * A + a0;
  for (int cc = wave; cc * 16 < ct; cc += 8) {
    int col = cc * 16 + lrow;
    // prefetch next chunk's weight column into the cache hierarchy
    int ncol = (cc + 8) * 16 + lrow;
    if (ncol < ct) __builtin_prefetch(WtT + (size_t)ncol * cs_p2, 0, 0);
    v8f c8 = {};
    if (accumulate) {
      #pragma unroll
      for (int j = 0; j < 8; ++j)
        c8[j] = acc[(accBase + j + 8 * klane) * ct + col];
    }
    for (int kk = 0; kk < nKC2; ++kk) {
      v16h af = load_frag_a(sS, cs_p2, lrow, kk * 32, klane);
      v16h bf = *(const v16h*)(WtT + (size_t)col * cs_p2 + kk * 32 + klane * 16);
      c8 = __builtin_amdgcn_wmma_f32_16x16x32_f16(false, af, false, bf,
                                                  (short)0, c8, false, false);
    }
    #pragma unroll
    for (int j = 0; j < 8; ++j)
      acc[(accBase + j + 8 * klane) * ct + col] = c8[j];
  }
}

// feats = relu((acc + bt) * gt + Bt), in place
__global__ void bn_relu_kernel(float* __restrict__ x, const float* __restrict__ bt,
                               const float* __restrict__ gt, const float* __restrict__ Bt,
                               int ct, size_t total) {
  size_t i = (size_t)blockIdx.x * blockDim.x + threadIdx.x;
  if (i >= total) return;
  int c = (int)(i % ct);
  float v = (x[i] + bt[c]) * gt[c] + Bt[c];
  x[i] = v > 0.f ? v : 0.f;
}

// WsT[col*din_p + k] = Ws[k][col] (f16, zero padded)
__global__ void convert_wsT_kernel(const float* __restrict__ W, _Float16* __restrict__ WT,
                                   int din, int cs, int din_p, int n) {
  int i = blockIdx.x * blockDim.x + threadIdx.x;
  if (i >= n) return;
  int col = i / din_p, k = i - col * din_p;
  float v = (k < din && col < cs) ? W[(size_t)k * cs + col] : 0.f;
  WT[i] = (_Float16)v;
}

// WtT[col*cs_p2 + k] = Wt[k][col] (f16, zero padded)
__global__ void convert_wtT_kernel(const float* __restrict__ W, _Float16* __restrict__ WT,
                                   int csK, int cs_p2, int ct, int n) {
  int i = blockIdx.x * blockDim.x + threadIdx.x;
  if (i >= n) return;
  int col = i / cs_p2, k = i - col * cs_p2;
  WT[i] = (_Float16)((k < csK) ? W[(size_t)k * ct + col] : 0.f);
}

// pooled[b][c] = max_t mean_a feats[b][t][a][c]
__global__ void pool_kernel(const float* __restrict__ f, float* __restrict__ pooled,
                            int T, int A, int C) {
  int b = blockIdx.x;
  for (int c = threadIdx.x; c < C; c += blockDim.x) {
    float best = -3.4e38f;
    for (int t = 0; t < T; ++t) {
      float s = 0.f;
      for (int a = 0; a < A; ++a) s += f[(((size_t)b * T + t) * A + a) * C + c];
      float m = s / (float)A;
      best = best > m ? best : m;
    }
    pooled[(size_t)b * C + c] = best;
  }
}

__global__ void fc_kernel(const float* __restrict__ p, const float* __restrict__ W,
                          const float* __restrict__ bias, float* __restrict__ out,
                          int B, int C, int O) {
  int i = blockIdx.x * blockDim.x + threadIdx.x;
  if (i >= B * O) return;
  int b = i / O, o = i - b * O;
  float s = bias[o];
  for (int k = 0; k < C; ++k) s += p[(size_t)b * C + k] * W[(size_t)k * O + o];
  out[i] = s;
}

struct LCfg { int cin, samp, cs, tr, ts, ct; float r; };
static const LCfg LC[6] = {
  {0,    2, 45,   0, 1, 64,   0.1f},
  {64,   2, 96,   1, 2, 128,  0.2f},
  {128,  1, 192,  1, 1, 256,  0.2f},
  {256,  2, 384,  1, 2, 512,  0.4f},
  {512,  1, 768,  1, 1, 1024, 0.4f},
  {1024, 2, 1536, 0, 1, 2048, 0.4f}
};
static inline int rup(int x, int m) { return (x + m - 1) / m * m; }

extern "C" void kernel_launch(void* const* d_in, const int* in_sizes, int n_in,
                              void* d_out, int out_size, void* d_ws, size_t ws_size,
                              hipStream_t stream) {
  (void)in_sizes; (void)n_in; (void)out_size; (void)ws_size;
  const int B = 4;
  const float* xyzs = (const float*)d_in[0];

  // bump allocator over workspace (256B aligned)
  char* wsp = (char*)d_ws;
  size_t off = 0;
  auto alloc = [&](size_t bytes) -> void* {
    off = (off + 255) & ~(size_t)255;
    void* p = wsp + off;
    off += bytes;
    return p;
  };

  float* ancBuf[2]  = { (float*)alloc((size_t)294912 * 4),  (float*)alloc((size_t)294912 * 4) };
  float* featBuf[2] = { (float*)alloc((size_t)6291456 * 4), (float*)alloc((size_t)6291456 * 4) };
  float* pooled     = (float*)alloc((size_t)B * 2048 * 4);

  (void)hipFuncSetAttribute(reinterpret_cast<const void*>(pst_fused_kernel),
                            hipFuncAttributeMaxDynamicSharedMemorySize, 98304);

  int T_in = 24, N = 2048, cur = 0;
  const float* xin = xyzs;
  const float* fin = nullptr;

  for (int i = 0; i < 6; ++i) {
    const LCfg& L = LC[i];
    const int A      = N / L.samp;
    const int T_out  = (T_in - 2 * L.tr + L.ts - 1) / L.ts;
    const int din    = 3 + L.cin;
    const int din_p  = rup(din, 32);
    const int cs_p16 = rup(L.cs, 16);
    const int cs_p2  = rup(L.cs, 32);
    const int W      = 2 * L.tr + 1;
    const int ct     = L.ct;

    const float* Ws = (const float*)d_in[1 + 8 * i + 0];
    const float* bs = (const float*)d_in[1 + 8 * i + 1];
    const float* gs = (const float*)d_in[1 + 8 * i + 2];
    const float* Bs = (const float*)d_in[1 + 8 * i + 3];
    const float* Wt = (const float*)d_in[1 + 8 * i + 4];
    const float* bt = (const float*)d_in[1 + 8 * i + 5];
    const float* gt = (const float*)d_in[1 + 8 * i + 6];
    const float* Bt = (const float*)d_in[1 + 8 * i + 7];

    _Float16* WsT = (_Float16*)alloc((size_t)cs_p16 * din_p * 2);
    _Float16* WtT = (_Float16*)alloc((size_t)W * ct * cs_p2 * 2);

    { int n = cs_p16 * din_p;
      convert_wsT_kernel<<<(n + 255) / 256, 256, 0, stream>>>(Ws, WsT, din, L.cs, din_p, n); }
    for (int w = 0; w < W; ++w) {
      int n = ct * cs_p2;
      convert_wtT_kernel<<<(n + 255) / 256, 256, 0, stream>>>(
          Wt + (size_t)w * L.cs * ct, WtT + (size_t)w * ct * cs_p2, L.cs, cs_p2, ct, n);
    }

    float* accb = featBuf[cur ^ 1];
    float* anco = ancBuf[cur ^ 1];
    const int smemB = N * 12 + 768 + 32 * (cs_p2 + din_p);

    for (int w = 0; w < W; ++w) {
      pst_fused_kernel<<<dim3(A / 16, T_out, B), 256, smemB, stream>>>(
          xin, fin, WsT, bs, gs, Bs, WtT + (size_t)w * ct * cs_p2, accb, anco,
          T_in, N, A, T_out, L.cin, L.cs, cs_p16, cs_p2, din_p, ct,
          L.tr, L.ts, L.samp, w - L.tr, L.r * L.r, (w > 0) ? 1 : 0);
    }

    size_t tot = (size_t)B * T_out * A * ct;
    bn_relu_kernel<<<(int)((tot + 255) / 256), 256, 0, stream>>>(accb, bt, gt, Bt, ct, tot);

    xin = anco; fin = accb; cur ^= 1; T_in = T_out; N = A;
  }

  pool_kernel<<<B, 256, 0, stream>>>(fin, pooled, T_in, N, 2048);
  fc_kernel<<<1, 256, 0, stream>>>(pooled, (const float*)d_in[49], (const float*)d_in[50],
                                   (float*)d_out, B, 2048, 60);
}
